// GroupedBatchedMLP_9457517986493
// MI455X (gfx1250) — matrix operand order repaired and checked by reading, not compiled, not run on hardware
//
#include <hip/hip_runtime.h>
#include <hip/hip_bf16.h>
#include <math.h>

#define NE 64
#define H  1024
#define F  2048
#define T  16384

#define TM 128
#define TN 128
#define TK 32
#define LDSB 112   // bytes per LDS tile row: 64B of bf16 data + 48B pad (28-bank stagger)

typedef __attribute__((ext_vector_type(16))) __bf16       v16bf;
typedef __attribute__((ext_vector_type(2)))  __bf16       v2bf;
typedef __attribute__((ext_vector_type(8)))  float        v8f;
typedef __attribute__((ext_vector_type(4)))  float        f32x4;
typedef __attribute__((ext_vector_type(4)))  unsigned int u32x4;

union FragAB {
  u32x4 q[2];
  v16bf v;
};

struct TileInfo { int expert; int rowStart; int rows; int pad; };

// Native conversions -> backend selects hardware bf16 cvt (pk form when possible)
__device__ __forceinline__ unsigned short f2bf(float f) {
  __bf16 b = (__bf16)f;
  return __builtin_bit_cast(unsigned short, b);
}

__device__ __forceinline__ unsigned int pack_bf16x2(float lo, float hi) {
  v2bf p;
  p.x = (__bf16)lo;
  p.y = (__bf16)hi;
  return __builtin_bit_cast(unsigned int, p);
}

__device__ __forceinline__ u32x4 cvt8(f32x4 a, f32x4 b) {
  u32x4 r;
  r.x = pack_bf16x2(a.x, a.y);
  r.y = pack_bf16x2(a.z, a.w);
  r.z = pack_bf16x2(b.x, b.y);
  r.w = pack_bf16x2(b.z, b.w);
  return r;
}

__device__ __forceinline__ u32x4 zero_u32x4() {
  u32x4 z; z.x = 0u; z.y = 0u; z.z = 0u; z.w = 0u; return z;
}

// One K=32 step: 8 waves cover a 128x128 tile; each wave = 32(M) x 64(N),
// i.e. 2x4 fragments of v_wmma_f32_16x16x32_bf16.
// A frag (16x32 bf16): lane L<16 -> row L, K {kh*8..kh*8+7, kh*8+16..+23}, kh=L>>4
// B frag (32x16 bf16): lane L    -> col L&15, K {kh*8..kh*8+7, kh*8+16..+23}, kh=L>>4
__device__ __forceinline__ void wmma_step(const char* sA, const char* sB,
                                          int waveM, int waveN, int lane,
                                          v8f acc[2][4]) {
  const int r  = lane & 15;
  const int kh = lane >> 4;
  FragAB a[2], b[4];
#pragma unroll
  for (int mi = 0; mi < 2; ++mi) {
    const char* base = sA + (waveM * 32 + mi * 16 + r) * LDSB + kh * 16;
    a[mi].q[0] = *(const u32x4*)(base);
    a[mi].q[1] = *(const u32x4*)(base + 32);
  }
#pragma unroll
  for (int ni = 0; ni < 4; ++ni) {
    const char* base = sB + (waveN * 64 + ni * 16 + r) * LDSB + kh * 16;
    b[ni].q[0] = *(const u32x4*)(base);
    b[ni].q[1] = *(const u32x4*)(base + 32);
  }
#pragma unroll
  for (int mi = 0; mi < 2; ++mi)
#pragma unroll
    for (int ni = 0; ni < 4; ++ni)
      acc[mi][ni] = __builtin_amdgcn_wmma_f32_16x16x32_bf16(
          false, a[mi].v, false, b[ni].v, (short)0, acc[mi][ni], false, false);
}

// Build tile map: each tile = up to 128 consecutive rows of one expert.
__global__ void moe_setup(const int* __restrict__ tpe, int* __restrict__ nTiles,
                          TileInfo* __restrict__ tiles) {
  if (threadIdx.x != 0 || blockIdx.x != 0) return;
  int off = 0, cnt = 0;
  for (int e = 0; e < NE; ++e) {
    int m = tpe[e];
    int s = off;
    off += m;
    while (m > 0) {
      int rows = m < TM ? m : TM;
      TileInfo ti; ti.expert = e; ti.rowStart = s; ti.rows = rows; ti.pad = 0;
      tiles[cnt++] = ti;
      s += rows; m -= rows;
    }
  }
  *nTiles = cnt;
}

// GEMM1: h = gelu(x @ w1[e]^T), both operands K(=H)-contiguous. Output bf16.
__global__ __launch_bounds__(256) void moe_gemm1(
    const float* __restrict__ x, const float* __restrict__ w1,
    const TileInfo* __restrict__ tiles, const int* __restrict__ nTiles,
    unsigned short* __restrict__ hbuf) {
  __shared__ __align__(16) char smem[2 * TM * LDSB];
  char* sA = smem;
  char* sB = smem + TM * LDSB;

  if ((int)blockIdx.x >= *nTiles) return;
  const TileInfo ti = tiles[blockIdx.x];
  const int rowsValid = ti.rows;
  const int rowStart  = ti.rowStart;
  const int nBase     = blockIdx.y * TN;   // over F

  const int tid   = threadIdx.x;
  const int lane  = tid & 31;
  const int wave  = tid >> 5;
  const int waveM = wave >> 1;
  const int waveN = wave & 1;

  const int lrow  = tid >> 1;   // 0..127: tile row (A) / tile col (B)
  const int lhalf = tid & 1;    // 16-float half of the K=32 slab

  const float* __restrict__ xrow = (lrow < rowsValid)
      ? (x + (size_t)(rowStart + lrow) * H + lhalf * 16) : (const float*)0;
  const float* __restrict__ w1row =
      w1 + ((size_t)ti.expert * F + (size_t)(nBase + lrow)) * H + lhalf * 16;

  v8f acc[2][4];
#pragma unroll
  for (int mi = 0; mi < 2; ++mi)
#pragma unroll
    for (int ni = 0; ni < 4; ++ni)
#pragma unroll
      for (int q = 0; q < 8; ++q) acc[mi][ni][q] = 0.0f;

  for (int k0 = 0; k0 < H; k0 += TK) {
    u32x4 pa0, pa1, pb0, pb1;
    if (xrow) {
      const f32x4* s = (const f32x4*)(xrow + k0);
      pa0 = cvt8(s[0], s[1]);
      pa1 = cvt8(s[2], s[3]);
    } else {
      pa0 = zero_u32x4();
      pa1 = zero_u32x4();
    }
    {
      const f32x4* s = (const f32x4*)(w1row + k0);
      pb0 = cvt8(s[0], s[1]);
      pb1 = cvt8(s[2], s[3]);
    }
    if (k0 + TK < H) __builtin_prefetch(w1row + k0 + TK, 0, 1);

    __syncthreads();   // previous step's LDS reads done
    *(u32x4*)(sA + lrow * LDSB + lhalf * 32)      = pa0;
    *(u32x4*)(sA + lrow * LDSB + lhalf * 32 + 16) = pa1;
    *(u32x4*)(sB + lrow * LDSB + lhalf * 32)      = pb0;
    *(u32x4*)(sB + lrow * LDSB + lhalf * 32 + 16) = pb1;
    __syncthreads();   // tile visible
    wmma_step(sA, sB, waveM, waveN, lane, acc);
  }

  // GELU (exact, erf) epilogue -> bf16 h
  const float ISQRT2 = 0.7071067811865476f;
#pragma unroll
  for (int mi = 0; mi < 2; ++mi) {
    const int mBase = waveM * 32 + mi * 16 + ((lane >> 4) << 3);
#pragma unroll
    for (int ni = 0; ni < 4; ++ni) {
      const int col = nBase + waveN * 64 + ni * 16 + (lane & 15);
#pragma unroll
      for (int j = 0; j < 8; ++j) {
        const int mrow = mBase + j;
        if (mrow < rowsValid) {
          float v = acc[mi][ni][j];
          float g = 0.5f * v * (1.0f + erff(v * ISQRT2));
          hbuf[(size_t)(rowStart + mrow) * F + col] = f2bf(g);
        }
      }
    }
  }
}

// GEMM2: out = h @ w2[e]. A is bf16 (K-contiguous); B (w2) is N-contiguous so
// the loader transposes fp32 k-pairs into packed bf16x2 dwords in LDS.
__global__ __launch_bounds__(256) void moe_gemm2(
    const unsigned short* __restrict__ hbuf, const float* __restrict__ w2,
    const TileInfo* __restrict__ tiles, const int* __restrict__ nTiles,
    float* __restrict__ out) {
  __shared__ __align__(16) char smem[2 * TM * LDSB];
  char* sA = smem;
  char* sB = smem + TM * LDSB;

  if ((int)blockIdx.x >= *nTiles) return;
  const TileInfo ti = tiles[blockIdx.x];
  const int rowsValid = ti.rows;
  const int rowStart  = ti.rowStart;
  const int nBase     = blockIdx.y * TN;   // over H

  const int tid   = threadIdx.x;
  const int lane  = tid & 31;
  const int wave  = tid >> 5;
  const int waveM = wave >> 1;
  const int waveN = wave & 1;

  // A loader
  const int lrow  = tid >> 1;
  const int lhalf = tid & 1;
  const unsigned short* __restrict__ arow = (lrow < rowsValid)
      ? (hbuf + (size_t)(rowStart + lrow) * F + lhalf * 16) : (const unsigned short*)0;

  // B loader: thread owns 2 consecutive k rows x 8 n columns
  const int k0l = (tid >> 4) * 2;   // 0,2,..,30
  const int n0  = (tid & 15) * 8;   // 0..120
  const float* __restrict__ brow =
      w2 + (size_t)ti.expert * F * H + (size_t)k0l * H + nBase + n0;

  v8f acc[2][4];
#pragma unroll
  for (int mi = 0; mi < 2; ++mi)
#pragma unroll
    for (int ni = 0; ni < 4; ++ni)
#pragma unroll
      for (int q = 0; q < 8; ++q) acc[mi][ni][q] = 0.0f;

  for (int k0 = 0; k0 < F; k0 += TK) {
    u32x4 pa0, pa1;
    if (arow) {
      const u32x4* s = (const u32x4*)(arow + k0);
      pa0 = s[0];
      pa1 = s[1];
    } else {
      pa0 = zero_u32x4();
      pa1 = zero_u32x4();
    }
    const float* p0 = brow + (size_t)k0 * H;
    const float* p1 = p0 + H;
    f32x4 r0a = ((const f32x4*)p0)[0], r0b = ((const f32x4*)p0)[1];
    f32x4 r1a = ((const f32x4*)p1)[0], r1b = ((const f32x4*)p1)[1];
    if (k0 + TK < F) __builtin_prefetch(p0 + (size_t)TK * H, 0, 1);

    __syncthreads();
    *(u32x4*)(sA + lrow * LDSB + lhalf * 32)      = pa0;
    *(u32x4*)(sA + lrow * LDSB + lhalf * 32 + 16) = pa1;
    {
      float c0[8] = {r0a.x, r0a.y, r0a.z, r0a.w, r0b.x, r0b.y, r0b.z, r0b.w};
      float c1[8] = {r1a.x, r1a.y, r1a.z, r1a.w, r1b.x, r1b.y, r1b.z, r1b.w};
#pragma unroll
      for (int i = 0; i < 8; ++i)
        *(unsigned int*)(sB + (n0 + i) * LDSB + k0l * 2) = pack_bf16x2(c0[i], c1[i]);
    }
    __syncthreads();
    wmma_step(sA, sB, waveM, waveN, lane, acc);
  }

#pragma unroll
  for (int mi = 0; mi < 2; ++mi) {
    const int mBase = waveM * 32 + mi * 16 + ((lane >> 4) << 3);
#pragma unroll
    for (int ni = 0; ni < 4; ++ni) {
      const int col = nBase + waveN * 64 + ni * 16 + (lane & 15);
#pragma unroll
      for (int j = 0; j < 8; ++j) {
        const int mrow = mBase + j;
        if (mrow < rowsValid)
          out[(size_t)(rowStart + mrow) * H + col] = acc[mi][ni][j];
      }
    }
  }
}

extern "C" void kernel_launch(void* const* d_in, const int* in_sizes, int n_in,
                              void* d_out, int out_size, void* d_ws, size_t ws_size,
                              hipStream_t stream) {
  const float* x   = (const float*)d_in[0];
  const float* w1  = (const float*)d_in[1];
  const float* w2  = (const float*)d_in[2];
  const int*   tpe = (const int*)d_in[3];
  float* out = (float*)d_out;

  char* ws = (char*)d_ws;
  int*            nTiles = (int*)ws;
  TileInfo*       tiles  = (TileInfo*)(ws + 256);          // <= 192 * 16 B
  unsigned short* hbuf   = (unsigned short*)(ws + 8192);   // T*F bf16 = 64 MB

  moe_setup<<<dim3(1, 1, 1), dim3(32, 1, 1), 0, stream>>>(tpe, nTiles, tiles);

  const int maxTiles = T / TM + NE;  // 192 upper bound; blocks early-exit
  moe_gemm1<<<dim3(maxTiles, F / TN, 1), dim3(256, 1, 1), 0, stream>>>(
      x, w1, tiles, nTiles, hbuf);
  moe_gemm2<<<dim3(maxTiles, H / TN, 1), dim3(256, 1, 1), 0, stream>>>(
      hbuf, w2, tiles, nTiles, out);
}
